// ToyMoe_34376918237954
// MI455X (gfx1250) — matrix operand order repaired and checked by reading, not compiled, not run on hardware
//
#include <hip/hip_runtime.h>
#include <math.h>

#define HID 2048
#define FDIM 4096
#define NTOK 8192
#define BM 32                     // tokens per MoE block (2 WMMA M-tiles)
#define THREADS_MOE 1024          // 32 waves
#define WAVES_MOE 32
#define FCHUNK 2048
#define NCHUNK 2
#define HS_STRIDE (FCHUNK + 8)    // bf16 elems, pad -> conflict-mitigated b128 reads
#define XG_ROWS (NTOK + 64)       // pad rows so tile tails never read OOB

typedef __bf16 bf16_t;
typedef bf16_t v2bf  __attribute__((ext_vector_type(2)));
typedef bf16_t v8bf  __attribute__((ext_vector_type(8)));
typedef bf16_t v16bf __attribute__((ext_vector_type(16)));
typedef float  v2f   __attribute__((ext_vector_type(2)));
typedef float  v8f   __attribute__((ext_vector_type(8)));

// ---- workspace layout (bytes) ----
#define OFF_CNT   0                                   // 2 x u32 (memset to 0 per launch)
#define OFF_IDX0  256
#define OFF_IDX1  (OFF_IDX0 + NTOK * 4)
#define OFF_XG    (OFF_IDX1 + NTOK * 4)               // routed bf16 x rows
#define W_ELEMS   ((size_t)FDIM * HID)                // 8388608 elems per matrix
#define OFF_W     (OFF_XG + (size_t)XG_ROWS * HID * 2)
// total ~101 MB

#define WMMA_BF16(A, B, C) \
  __builtin_amdgcn_wmma_f32_16x16x32_bf16(false, (A), false, (B), (short)0, (C), false, false)

static __device__ inline v16bf load_afrag(const bf16_t* p) {
  // A fragment: 16 contiguous bf16 (k..k+15 for this lane half)
  v8bf lo = *(const v8bf*)p;
  v8bf hi = *(const v8bf*)(p + 8);
  return __builtin_shufflevector(lo, hi, 0, 1, 2, 3, 4, 5, 6, 7, 8, 9, 10, 11, 12, 13, 14, 15);
}

static __device__ inline float gelu_exact(float v) {
  return 0.5f * v * (1.0f + erff(v * 0.70710678118654752440f));
}

// ---------- K0: convert 4 weight matrices fp32 -> bf16 into ws ----------
__global__ __launch_bounds__(256)
void convert_w_kernel(const float* __restrict__ w0a, const float* __restrict__ w0b,
                      const float* __restrict__ w1a, const float* __restrict__ w1b,
                      bf16_t* __restrict__ dst) {
  const unsigned blk = blockIdx.x;          // 4 * 2048 blocks
  const unsigned seg = blk >> 11;
  const unsigned within = blk & 2047;
  const float* src = (seg == 0) ? w0a : (seg == 1) ? w0b : (seg == 2) ? w1a : w1b;
  bf16_t* d = dst + (size_t)seg * W_ELEMS;
  #pragma unroll
  for (int k = 0; k < 8; ++k) {             // 2048 pairs per block
    size_t p = (size_t)within * 2048 + threadIdx.x + (size_t)k * 256;
    v2f v = *(const v2f*)(src + 2 * p);
    v2bf o = {(bf16_t)v.x, (bf16_t)v.y};
    *(v2bf*)(d + 2 * p) = o;
  }
}

// ---------- K1: gate (argmax of logits == argmax of softmax) + route ----------
__global__ __launch_bounds__(256)
void gate_kernel(const float* __restrict__ x, const float* __restrict__ Wg,
                 unsigned* __restrict__ counts,
                 unsigned* __restrict__ idx0, unsigned* __restrict__ idx1) {
  const int lane = threadIdx.x & 31;
  const int wave = threadIdx.x >> 5;
  const int tok  = blockIdx.x * 8 + wave;   // 1024 blocks x 8 waves
  const float* xr = x + (size_t)tok * HID;
  float d0 = 0.f, d1 = 0.f;
  for (int k = lane; k < HID; k += 32) {
    float v = xr[k];
    d0 = fmaf(v, Wg[k], d0);
    d1 = fmaf(v, Wg[HID + k], d1);
  }
  #pragma unroll
  for (int off = 16; off > 0; off >>= 1) {
    d0 += __shfl_xor(d0, off, 32);
    d1 += __shfl_xor(d1, off, 32);
  }
  if (lane == 0) {
    int e = (d1 > d0) ? 1 : 0;              // tie -> expert 0 (matches argmax)
    unsigned pos = atomicAdd(&counts[e], 1u);
    (e ? idx1 : idx0)[pos] = (unsigned)tok;
  }
}

// ---------- K2: gather x rows into routed order, fp32 -> bf16 ----------
__global__ __launch_bounds__(256)
void gather_x_kernel(const float* __restrict__ x, const unsigned* __restrict__ counts,
                     const unsigned* __restrict__ idx0, const unsigned* __restrict__ idx1,
                     bf16_t* __restrict__ xg) {
  const unsigned g  = blockIdx.x;           // 8192 routed rows
  const unsigned c0 = counts[0];
  const unsigned tok = (g < c0) ? idx0[g] : idx1[g - c0];
  const float* src = x + (size_t)tok * HID;
  bf16_t* dst = xg + (size_t)g * HID;
  #pragma unroll
  for (int it = 0; it < 4; ++it) {
    int colp = threadIdx.x + it * 256;      // pair index
    v2f v = *(const v2f*)(src + 2 * colp);
    v2bf o = {(bf16_t)v.x, (bf16_t)v.y};
    *(v2bf*)(dst + 2 * colp) = o;
  }
}

// ---------- K3: routed fused expert MLP, bf16 WMMA, fp32 accumulate ----------
__global__ __launch_bounds__(THREADS_MOE)
void moe_gemm_kernel(const bf16_t* __restrict__ xg, const unsigned* __restrict__ counts,
                     const unsigned* __restrict__ idx0, const unsigned* __restrict__ idx1,
                     const bf16_t* __restrict__ wbf,   // [W0a,W0b,W1a,W1b] bf16
                     const float* __restrict__ b0a, const float* __restrict__ b0b,
                     const float* __restrict__ b1a, const float* __restrict__ b1b,
                     float* __restrict__ out) {
  extern __shared__ char smem_raw[];
  bf16_t* hs = (bf16_t*)smem_raw;           // [BM][HS_STRIDE] activated h chunk (bf16)
  __shared__ int s_tok[BM];

  const unsigned c0 = counts[0];
  const unsigned c1 = counts[1];
  const unsigned B0 = (c0 + BM - 1) / BM;
  const unsigned B1 = (c1 + BM - 1) / BM;
  const unsigned b  = blockIdx.x;
  if (b >= B0 + B1) return;

  int e; unsigned tile, cnt, rowbase; const unsigned* list;
  if (b < B0) { e = 0; tile = b;      cnt = c0; rowbase = tile * BM;      list = idx0; }
  else        { e = 1; tile = b - B0; cnt = c1; rowbase = c0 + tile * BM; list = idx1; }

  const bf16_t* Wa = wbf + (size_t)(2 * e) * W_ELEMS;
  const bf16_t* Wb = wbf + (size_t)(2 * e + 1) * W_ELEMS;
  const float*  ba = e ? b1a : b0a;
  const float*  bb = e ? b1b : b0b;

  const int tid  = threadIdx.x;
  const int lane = tid & 31;
  const int wave = tid >> 5;
  const int hi   = lane >> 4;
  const int lm   = lane & 15;

  if (tid < BM) {
    unsigned pos = tile * BM + (unsigned)tid;
    s_tok[tid] = (pos < cnt) ? (int)list[pos] : -1;   // -1 => masked store
  }

  const int jb = wave * (HID / WAVES_MOE);            // 64 out cols per wave
  const v8f vzero = (v8f){0.f, 0.f, 0.f, 0.f, 0.f, 0.f, 0.f, 0.f};
  v8f acc[8];                                          // [Mtile(2)][Ntile(4)]
  #pragma unroll
  for (int i = 0; i < 8; ++i) acc[i] = vzero;

  const bf16_t* a0base = xg + (size_t)(rowbase + lm) * HID + 16 * hi;
  const bf16_t* a1base = a0base + (size_t)16 * HID;

  #pragma unroll 1
  for (int c = 0; c < NCHUNK; ++c) {
    const int f0 = c * FCHUNK;
    const int nb = wave * (FCHUNK / WAVES_MOE);       // 64 h cols per wave per chunk

    // ---- up-projection: h_chunk = act(x @ Wa^T + ba), bf16 into LDS ----
    // two N-tiles per pass -> 4 independent WMMA chains (hides 5-slot WMMA RAW hazard)
    #pragma unroll 1
    for (int tp = 0; tp < 2; ++tp) {
      const int fcA = nb + (2 * tp) * 16;             // col (within chunk) of tile A
      const int fcB = fcA + 16;                       // col of tile B
      const bf16_t* bbA = Wa + (size_t)(f0 + fcA + lm) * HID + 16 * hi;
      const bf16_t* bbB = Wa + (size_t)(f0 + fcB + lm) * HID + 16 * hi;
      v8f h00 = vzero, h10 = vzero, h01 = vzero, h11 = vzero;
      #pragma unroll 2
      for (int kk = 0; kk < HID; kk += 32) {
        v16bf bfA = *(const v16bf*)(bbA + kk);
        v16bf bfB = *(const v16bf*)(bbB + kk);
        v16bf a0  = load_afrag(a0base + kk);
        v16bf a1  = load_afrag(a1base + kk);
        h00 = WMMA_BF16(a0, bfA, h00);                // A pair shared by 4 WMMAs,
        h10 = WMMA_BF16(a1, bfA, h10);                // each B shared by 2 M-tiles
        h01 = WMMA_BF16(a0, bfB, h01);
        h11 = WMMA_BF16(a1, bfB, h11);
      }
      const float bvA = ba[f0 + fcA + lm];
      const float bvB = ba[f0 + fcB + lm];
      const int colA = fcA + lm;
      const int colB = fcB + lm;
      if (e == 0) {
        #pragma unroll
        for (int r = 0; r < 8; ++r) {
          hs[(r + 8 * hi) * HS_STRIDE + colA]      = (bf16_t)gelu_exact(h00[r] + bvA);
          hs[(16 + r + 8 * hi) * HS_STRIDE + colA] = (bf16_t)gelu_exact(h10[r] + bvA);
          hs[(r + 8 * hi) * HS_STRIDE + colB]      = (bf16_t)gelu_exact(h01[r] + bvB);
          hs[(16 + r + 8 * hi) * HS_STRIDE + colB] = (bf16_t)gelu_exact(h11[r] + bvB);
        }
      } else {
        #pragma unroll
        for (int r = 0; r < 8; ++r) {
          hs[(r + 8 * hi) * HS_STRIDE + colA]      = (bf16_t)fmaxf(h00[r] + bvA, 0.f);
          hs[(16 + r + 8 * hi) * HS_STRIDE + colA] = (bf16_t)fmaxf(h10[r] + bvA, 0.f);
          hs[(r + 8 * hi) * HS_STRIDE + colB]      = (bf16_t)fmaxf(h01[r] + bvB, 0.f);
          hs[(16 + r + 8 * hi) * HS_STRIDE + colB] = (bf16_t)fmaxf(h11[r] + bvB, 0.f);
        }
      }
    }
    __syncthreads();

    // ---- down-projection accumulate: acc += h_chunk @ Wb^T ----
    // 8 accumulator chains -> no WMMA RAW stalls; A (LDS) reused across 4 N-tiles
    const bf16_t* hA0 = hs + (size_t)lm * HS_STRIDE + 16 * hi;
    const bf16_t* hA1 = hs + (size_t)(16 + lm) * HS_STRIDE + 16 * hi;
    #pragma unroll 2
    for (int kk = 0; kk < FCHUNK; kk += 32) {
      v16bf a0 = load_afrag(hA0 + kk);
      v16bf a1 = load_afrag(hA1 + kk);
      #pragma unroll
      for (int t = 0; t < 4; ++t) {
        v16bf bf = *(const v16bf*)(Wb + (size_t)(jb + t * 16 + lm) * FDIM + f0 + 16 * hi + kk);
        acc[t]     = WMMA_BF16(a0, bf, acc[t]);       // B shared by both M-tiles
        acc[4 + t] = WMMA_BF16(a1, bf, acc[4 + t]);
      }
    }
    __syncthreads();                                  // hs reused next chunk
  }

  // ---- scatter results back to token rows (masked tails) ----
  #pragma unroll 1
  for (int t = 0; t < 4; ++t) {
    const int j = jb + t * 16 + lm;
    const float bv = bb[j];
    #pragma unroll
    for (int r = 0; r < 8; ++r) {
      const int tok0 = s_tok[r + 8 * hi];
      if (tok0 >= 0) out[(size_t)tok0 * HID + j] = acc[t][r] + bv;
      const int tok1 = s_tok[16 + r + 8 * hi];
      if (tok1 >= 0) out[(size_t)tok1 * HID + j] = acc[4 + t][r] + bv;
    }
  }
}

extern "C" void kernel_launch(void* const* d_in, const int* in_sizes, int n_in,
                              void* d_out, int out_size, void* d_ws, size_t ws_size,
                              hipStream_t stream) {
  (void)in_sizes; (void)n_in; (void)out_size; (void)ws_size;
  const float* x   = (const float*)d_in[0];
  const float* Wg  = (const float*)d_in[1];
  const float* W0a = (const float*)d_in[2];
  const float* b0a = (const float*)d_in[3];
  const float* W0b = (const float*)d_in[4];
  const float* b0b = (const float*)d_in[5];
  const float* W1a = (const float*)d_in[6];
  const float* b1a = (const float*)d_in[7];
  const float* W1b = (const float*)d_in[8];
  const float* b1b = (const float*)d_in[9];
  float* out = (float*)d_out;

  char* ws = (char*)d_ws;
  unsigned* counts = (unsigned*)(ws + OFF_CNT);
  unsigned* idx0   = (unsigned*)(ws + OFF_IDX0);
  unsigned* idx1   = (unsigned*)(ws + OFF_IDX1);
  bf16_t*   xg     = (bf16_t*)(ws + OFF_XG);
  bf16_t*   wbf    = (bf16_t*)(ws + OFF_W);

  hipMemsetAsync(counts, 0, 2 * sizeof(unsigned), stream);

  convert_w_kernel<<<4 * 2048, 256, 0, stream>>>(W0a, W0b, W1a, W1b, wbf);
  gate_kernel<<<NTOK / 8, 256, 0, stream>>>(x, Wg, counts, idx0, idx1);
  gather_x_kernel<<<NTOK, 256, 0, stream>>>(x, counts, idx0, idx1, xg);

  const size_t shmem = (size_t)BM * HS_STRIDE * sizeof(bf16_t);   // ~128.5 KB
  (void)hipFuncSetAttribute((const void*)moe_gemm_kernel,
                            hipFuncAttributeMaxDynamicSharedMemorySize, (int)shmem);
  // worst-case blocks: ceil(c0/32)+ceil(c1/32) <= 257
  moe_gemm_kernel<<<258, THREADS_MOE, shmem, stream>>>(
      xg, counts, idx0, idx1, wbf, b0a, b0b, b1a, b1b, out);
}
